// SelfAttention_41729902248320
// MI455X (gfx1250) — compile-verified, matrix operand order checked by reading, not run on hardware
//
#include <hip/hip_runtime.h>
#include <hip/hip_bf16.h>
#include <math.h>

typedef __bf16 bf16_t;
typedef __attribute__((ext_vector_type(16))) __bf16 v16bf;
typedef __attribute__((ext_vector_type(8)))  __bf16 v8bf;
typedef __attribute__((ext_vector_type(8)))  float  v8f;

#define BATCH  4
#define SEQ    2048
#define DMODEL 512
#define HEADS  8
#define CDIM   64
#define MTOT   (BATCH*SEQ)   /* 8192 */

// ---------------------------------------------------------------------------
// Fragment loader for 16-bit WMMA operands (A 16x32 or B 32x16, both stored so
// that one lane owns one row/col with two contiguous 8-element K-chunks).
// Per ISA 16-bit layout: lanes 0-15 hold K-chunks {0..7, 16..23},
// lanes 16-31 hold {8..15, 24..31}  ->  koff = (lane>>4)*8, second chunk +16.
// ---------------------------------------------------------------------------
__device__ __forceinline__ v16bf frag_load(const bf16_t* rowptr, int koff) {
  v8bf lo = *(const v8bf*)(rowptr + koff);
  v8bf hi = *(const v8bf*)(rowptr + koff + 16);
  v16bf r;
#pragma unroll
  for (int i = 0; i < 8; ++i) { r[i] = lo[i]; r[i + 8] = hi[i]; }
  return r;
}

__device__ __forceinline__ v8f wmma_bf16(v16bf a, v16bf b, v8f c) {
  return __builtin_amdgcn_wmma_f32_16x16x32_bf16(
      /*neg_a=*/false, a, /*neg_b=*/false, b,
      /*c_mod=*/(short)0, c, /*reuse_a=*/false, /*reuse_b=*/false);
}

// ---------------------------------------------------------------------------
// Prep: fp32 -> bf16 elementwise
// ---------------------------------------------------------------------------
__global__ void __launch_bounds__(256)
cvt_f32_bf16(const float* __restrict__ in, bf16_t* __restrict__ out, int n) {
  int i = blockIdx.x * 256 + threadIdx.x;
  if (i < n) out[i] = (bf16_t)in[i];
}

// Prep: W [K=512][N=512] fp32 -> WT [N][K] bf16 (transpose+convert)
__global__ void __launch_bounds__(256)
transpose_w(const float* __restrict__ W, bf16_t* __restrict__ WT) {
  int i = blockIdx.x * 256 + threadIdx.x;   // i = r*512 + k
  int r = i >> 9, k = i & 511;
  WT[i] = (bf16_t)W[(size_t)k * DMODEL + r];
}

// ---------------------------------------------------------------------------
// Generic bf16 WMMA GEMM:  Out = A[M=8192,K=512] * Bt^T  (Bt is [N][K] bf16)
// One wave computes a 16x64 tile (A-fragment reused across 4 WMMAs per k-step).
// mode 0: bf16 row-major out (*scale)           (Q / K projections)
// mode 1: bf16 out transposed per head -> Vt[B][H][C][SEQ]   (V projection)
// mode 2: f32 row-major out + bias              (output projection)
// ---------------------------------------------------------------------------
__global__ void __launch_bounds__(256)
gemm_bf16(const bf16_t* __restrict__ A, const bf16_t* __restrict__ Bt,
          void* __restrict__ Out, const float* __restrict__ bias,
          float scale, int mode) {
  const int K = DMODEL, Nn = DMODEL;
  int lane   = threadIdx.x & 31;
  int wave   = threadIdx.x >> 5;
  int wid    = blockIdx.x * 8 + wave;
  int nb     = wid & 7;          // 8 column blocks of 64
  int mt     = wid >> 3;         // 512 row tiles of 16
  int ln     = lane & 15;
  int koff   = (lane >> 4) * 8;
  int rowoff = (lane >> 4) * 8;  // D-fragment: vgpr r -> row rowoff + r

  v8f acc[4];
#pragma unroll
  for (int nt = 0; nt < 4; ++nt)
#pragma unroll
    for (int i = 0; i < 8; ++i) acc[nt][i] = 0.f;

  const bf16_t* ap = A + (size_t)(mt * 16 + ln) * K;
  for (int k0 = 0; k0 < K; k0 += 32) {
    __builtin_prefetch(ap + k0 + 128, 0, 1);          // global_prefetch_b8
    v16bf a = frag_load(ap + k0, koff);
#pragma unroll
    for (int nt = 0; nt < 4; ++nt) {
      int col = nb * 64 + nt * 16 + ln;
      v16bf b = frag_load(Bt + (size_t)col * K + k0, koff);
      acc[nt] = wmma_bf16(a, b, acc[nt]);
    }
  }

  if (mode == 0) {
    bf16_t* O = (bf16_t*)Out;
#pragma unroll
    for (int nt = 0; nt < 4; ++nt) {
      int col = nb * 64 + nt * 16 + ln;
#pragma unroll
      for (int r = 0; r < 8; ++r)
        O[(size_t)(mt * 16 + rowoff + r) * Nn + col] = (bf16_t)(acc[nt][r] * scale);
    }
  } else if (mode == 1) {
    bf16_t* O = (bf16_t*)Out;  // Vt [B][H][C][SEQ]
#pragma unroll
    for (int nt = 0; nt < 4; ++nt) {
      int col = nb * 64 + nt * 16 + ln;
      int h = col >> 6, cc = col & 63;
#pragma unroll
      for (int r = 0; r < 8; ++r) {
        int t = mt * 16 + rowoff + r;          // global token
        int bi = t >> 11, i = t & (SEQ - 1);   // tile never crosses batch
        O[(size_t)((bi * HEADS + h) * CDIM + cc) * SEQ + i] =
            (bf16_t)(acc[nt][r] * scale);
      }
    }
  } else {
    float* O = (float*)Out;
#pragma unroll
    for (int nt = 0; nt < 4; ++nt) {
      int col = nb * 64 + nt * 16 + ln;
      float bv = bias[col];
#pragma unroll
      for (int r = 0; r < 8; ++r)
        O[(size_t)(mt * 16 + rowoff + r) * Nn + col] = acc[nt][r] * scale + bv;
    }
  }
}

// ---------------------------------------------------------------------------
// Fused flash attention: one wave per 16-query tile per (b,h).
// Qb/Kb: bf16 [8192][512] (Q pre-scaled by C^-0.5). Vt: bf16 [B][H][C][SEQ].
// Ob: bf16 [8192][512].
// ---------------------------------------------------------------------------
__global__ void __launch_bounds__(128)
attn_fused(const bf16_t* __restrict__ Qb, const bf16_t* __restrict__ Kb,
           const bf16_t* __restrict__ Vt, bf16_t* __restrict__ Ob) {
  __shared__ __align__(16) bf16_t Pl[4][16 * 32];   // per-wave P staging
  int lane   = threadIdx.x & 31;
  int wave   = threadIdx.x >> 5;
  int wg     = blockIdx.x * 4 + wave;
  int bh     = wg >> 7;                // (b,h) pair, 32 total
  int qt     = wg & 127;               // query tile within sequence
  int b      = bh >> 3, h = bh & 7;
  int ln     = lane & 15;
  int koff   = (lane >> 4) * 8;
  int rowoff = (lane >> 4) * 8;

  const bf16_t* qp = Qb + (size_t)(b * SEQ + qt * 16 + ln) * DMODEL + h * CDIM;
  v16bf qf0 = frag_load(qp, koff);        // c = 0..31
  v16bf qf1 = frag_load(qp + 32, koff);   // c = 32..63

  const bf16_t* Kbase = Kb + (size_t)(b * SEQ) * DMODEL + h * CDIM;
  const bf16_t* Vbase = Vt + (size_t)bh * CDIM * SEQ;
  bf16_t* myP = &Pl[wave][0];

  v8f o[4];
#pragma unroll
  for (int ct = 0; ct < 4; ++ct)
#pragma unroll
    for (int i = 0; i < 8; ++i) o[ct][i] = 0.f;
  float mrow[8], lrow[8];
#pragma unroll
  for (int r = 0; r < 8; ++r) { mrow[r] = -3.0e38f; lrow[r] = 0.f; }

  for (int j0 = 0; j0 < SEQ; j0 += 32) {
    v8f s0, s1;
#pragma unroll
    for (int i = 0; i < 8; ++i) { s0[i] = 0.f; s1[i] = 0.f; }
    {
      const bf16_t* kp = Kbase + (size_t)(j0 + ln) * DMODEL;   // keys j0..j0+15
      s0 = wmma_bf16(qf0, frag_load(kp, koff), s0);
      s0 = wmma_bf16(qf1, frag_load(kp + 32, koff), s0);
      kp += (size_t)16 * DMODEL;                               // keys j0+16..j0+31
      s1 = wmma_bf16(qf0, frag_load(kp, koff), s1);
      s1 = wmma_bf16(qf1, frag_load(kp + 32, koff), s1);
    }
    // Online softmax. D-fragment: vgpr r holds row (rowoff+r), column ln;
    // a row's 16 columns live across one 16-lane half -> xor-shuffle reduce.
#pragma unroll
    for (int r = 0; r < 8; ++r) {
      float v = fmaxf(s0[r], s1[r]);
      v = fmaxf(v, __shfl_xor(v, 1, 32));
      v = fmaxf(v, __shfl_xor(v, 2, 32));
      v = fmaxf(v, __shfl_xor(v, 4, 32));
      v = fmaxf(v, __shfl_xor(v, 8, 32));
      float mnew  = fmaxf(mrow[r], v);
      float alpha = __expf(mrow[r] - mnew);
      mrow[r] = mnew;
      float p0 = __expf(s0[r] - mnew);
      float p1 = __expf(s1[r] - mnew);
      float ps = p0 + p1;
      ps += __shfl_xor(ps, 1, 32);
      ps += __shfl_xor(ps, 2, 32);
      ps += __shfl_xor(ps, 4, 32);
      ps += __shfl_xor(ps, 8, 32);
      lrow[r] = lrow[r] * alpha + ps;
      o[0][r] *= alpha; o[1][r] *= alpha; o[2][r] *= alpha; o[3][r] *= alpha;
      myP[(rowoff + r) * 32 + ln]      = (bf16_t)p0;   // D-layout -> row major LDS
      myP[(rowoff + r) * 32 + 16 + ln] = (bf16_t)p1;
    }
    // Same-wave DS ops are in-order; wait for stores, then re-read in A-layout.
    asm volatile("s_wait_dscnt 0" ::: "memory");
    v16bf pf = frag_load(myP + ln * 32, koff);   // A-fragment of P (16x32)
#pragma unroll
    for (int ct = 0; ct < 4; ++ct) {
      const bf16_t* vp = Vbase + (size_t)(ct * 16 + ln) * SEQ + j0;
      o[ct] = wmma_bf16(pf, frag_load(vp, koff), o[ct]);
    }
  }

#pragma unroll
  for (int r = 0; r < 8; ++r) {
    float inv = 1.0f / lrow[r];
    o[0][r] *= inv; o[1][r] *= inv; o[2][r] *= inv; o[3][r] *= inv;
  }
  bf16_t* op = Ob + (size_t)(b * SEQ + qt * 16) * DMODEL + h * CDIM;
#pragma unroll
  for (int ct = 0; ct < 4; ++ct)
#pragma unroll
    for (int r = 0; r < 8; ++r)
      op[(size_t)(rowoff + r) * DMODEL + ct * 16 + ln] = (bf16_t)o[ct][r];
}

// ---------------------------------------------------------------------------
extern "C" void kernel_launch(void* const* d_in, const int* in_sizes, int n_in,
                              void* d_out, int out_size, void* d_ws, size_t ws_size,
                              hipStream_t stream) {
  const float* x  = (const float*)d_in[0];
  const float* Wq = (const float*)d_in[1];
  const float* Wk = (const float*)d_in[2];
  const float* Wv = (const float*)d_in[3];
  const float* Wo = (const float*)d_in[4];
  const float* bo = (const float*)d_in[5];
  float* out = (float*)d_out;

  char* ws = (char*)d_ws;
  const size_t SZ_ACT = (size_t)MTOT * DMODEL * sizeof(bf16_t);   // 8 MiB
  const size_t SZ_W   = (size_t)DMODEL * DMODEL * sizeof(bf16_t); // 512 KiB
  size_t off = 0;
  bf16_t* Xb  = (bf16_t*)(ws + off); off += SZ_ACT;
  bf16_t* WqT = (bf16_t*)(ws + off); off += SZ_W;
  bf16_t* WkT = (bf16_t*)(ws + off); off += SZ_W;
  bf16_t* WvT = (bf16_t*)(ws + off); off += SZ_W;
  bf16_t* WoT = (bf16_t*)(ws + off); off += SZ_W;
  bf16_t* Qb  = (bf16_t*)(ws + off); off += SZ_ACT;
  bf16_t* Kb  = (bf16_t*)(ws + off); off += SZ_ACT;
  bf16_t* Vt  = (bf16_t*)(ws + off); off += SZ_ACT;
  bf16_t* Ob  = (bf16_t*)(ws + off); off += SZ_ACT;

  // 1) convert inputs to bf16 (weights transposed for contiguous B-fragments)
  cvt_f32_bf16<<<(MTOT * DMODEL) / 256, 256, 0, stream>>>(x, Xb, MTOT * DMODEL);
  transpose_w<<<(DMODEL * DMODEL) / 256, 256, 0, stream>>>(Wq, WqT);
  transpose_w<<<(DMODEL * DMODEL) / 256, 256, 0, stream>>>(Wk, WkT);
  transpose_w<<<(DMODEL * DMODEL) / 256, 256, 0, stream>>>(Wv, WvT);
  transpose_w<<<(DMODEL * DMODEL) / 256, 256, 0, stream>>>(Wo, WoT);

  // 2) QKV projections (Q pre-scaled by C^-0.5 = 0.125; V stored transposed)
  const int GEMM_BLOCKS = (MTOT / 16) * (DMODEL / 64) / 8;  // 512
  gemm_bf16<<<GEMM_BLOCKS, 256, 0, stream>>>(Xb, WqT, Qb, nullptr, 0.125f, 0);
  gemm_bf16<<<GEMM_BLOCKS, 256, 0, stream>>>(Xb, WkT, Kb, nullptr, 1.0f, 0);
  gemm_bf16<<<GEMM_BLOCKS, 256, 0, stream>>>(Xb, WvT, Vt, nullptr, 1.0f, 1);

  // 3) fused flash attention: 4096 waves (32 bh pairs x 128 q-tiles), 4/block
  attn_fused<<<1024, 128, 0, stream>>>(Qb, Kb, Vt, Ob);

  // 4) output projection with bias, fp32 out
  gemm_bf16<<<GEMM_BLOCKS, 256, 0, stream>>>(Ob, WoT, out, bo, 1.0f, 2);
}
